// FFTAttention_67508295958990
// MI455X (gfx1250) — matrix operand order repaired and checked by reading, not compiled, not run on hardware
//
#include <hip/hip_runtime.h>

// ---------------------------------------------------------------------------
// FFT attention for MI455X (gfx1250).  All matmuls on v_wmma_f32_16x16x32_bf16.
// B=2, N=1024, D=1024, H=16, DH=64.
// ---------------------------------------------------------------------------

typedef __bf16 bf16;
typedef __attribute__((ext_vector_type(16))) __bf16 v16bf;
typedef __attribute__((ext_vector_type(8)))  float  v8f;

union FragAB { v16bf v; unsigned int u[8]; };

#define WMMA_BF16(a, b, c) \
  __builtin_amdgcn_wmma_f32_16x16x32_bf16(false, (a), false, (b), (short)0, (c), false, false)

enum { kB = 2, kN = 1024, kD = 1024, kH = 16, kDH = 64, kM = kB * kN };
enum { kBHND = kB * kH * kN * kDH };   // 2,097,152 elements per planar tensor

static __device__ __forceinline__ v8f v8f_zero() {
  v8f z;
#pragma unroll
  for (int i = 0; i < 8; ++i) z[i] = 0.0f;
  return z;
}

// ---------------------------------------------------------------------------
// Elementwise fp32 -> bf16
// ---------------------------------------------------------------------------
__global__ __launch_bounds__(256) void k_cvt_bf16(const float* __restrict__ src,
                                                  bf16* __restrict__ dst, int n) {
  int i = blockIdx.x * 256 + threadIdx.x;
  if (i < n) dst[i] = (bf16)src[i];
}

// vf planar [BH][N][DH] fp32 -> transposed [BH][DH][N] bf16 (B-fragment friendly)
__global__ __launch_bounds__(256) void k_cvt_vfT(const float* __restrict__ src,
                                                 bf16* __restrict__ dst) {
  int i = blockIdx.x * 256 + threadIdx.x;          // over kBHND
  int bh = i >> 16;                                // / (N*DH)
  int rem = i & 65535;
  int n = rem >> 6;
  int d = rem & 63;
  dst[bh * 65536 + d * kN + n] = (bf16)src[i];
}

// out_t planar [b][h][n][dh] fp32 -> A matrix [b*N + n][h*DH + dh] bf16
__global__ __launch_bounds__(256) void k_cvt_outT(const float* __restrict__ src,
                                                  bf16* __restrict__ dst) {
  int i = blockIdx.x * 256 + threadIdx.x;          // over kBHND
  int b = i >> 20;
  int rem = i & 1048575;
  int h = rem >> 16;
  int rem2 = rem & 65535;
  int n = rem2 >> 6;
  int dh = rem2 & 63;
  dst[(size_t)((b << 10) + n) * kD + (h << 6) + dh] = (bf16)src[i];
}

// ---------------------------------------------------------------------------
// 1024-point radix-2 complex FFT (ortho), one transform per workgroup.
// Planar re/im arrays laid out [BH][N][DH]; element n strided by DH.
// sgn = -1 forward, +1 inverse.
// ---------------------------------------------------------------------------
__global__ __launch_bounds__(256) void k_fft1024(float* __restrict__ re,
                                                 float* __restrict__ im, float sgn) {
  __shared__ float sr[1024];
  __shared__ float si[1024];
  int t = threadIdx.x;
  int wg = blockIdx.x;                 // 0 .. B*H*DH-1
  int bh = wg >> 6;
  int dh = wg & 63;
  size_t base = (size_t)bh * 65536 + dh;

#pragma unroll
  for (int e = 0; e < 4; ++e) {
    int n = e * 256 + t;
    int j = (int)(__brev((unsigned)n) >> 22);      // 10-bit reversal
    sr[j] = re[base + (size_t)n * 64];
    si[j] = im[base + (size_t)n * 64];
  }
  __syncthreads();

  for (int s = 1; s <= 10; ++s) {
    int half = 1 << (s - 1);
#pragma unroll
    for (int e = 0; e < 2; ++e) {
      int bt = e * 256 + t;
      int grp = bt >> (s - 1);
      int pos = bt & (half - 1);
      int i0 = (grp << s) + pos;
      int i1 = i0 + half;
      float ang = sgn * 6.2831853071795864f * (float)pos / (float)(half * 2);
      float sn, cs;
      __sincosf(ang, &sn, &cs);
      float xr = sr[i1], xi = si[i1];
      float tr = cs * xr - sn * xi;
      float ti = cs * xi + sn * xr;
      float ar = sr[i0], ai = si[i0];
      sr[i1] = ar - tr;  si[i1] = ai - ti;
      sr[i0] = ar + tr;  si[i0] = ai + ti;
    }
    __syncthreads();
  }

#pragma unroll
  for (int e = 0; e < 4; ++e) {
    int n = e * 256 + t;
    re[base + (size_t)n * 64] = sr[n] * 0.03125f;   // 1/sqrt(1024)
    im[base + (size_t)n * 64] = si[n] * 0.03125f;
  }
}

// ---------------------------------------------------------------------------
// bf16 WMMA GEMM:  C[M,Nd] = A1[M,K] @ W1[Nd,K]^T (+/- A2 @ W2^T) + bias
// Block tile 128x128, K-step 32, 8 waves (4x2), 2x4 16x16 WMMA tiles per wave.
// Double-buffered LDS; next tile register-staged during current compute; all
// fragments loaded before the 8 back-to-back WMMAs (one dscnt wait per step).
// mode 0: scatter fp32 into [B,H,N,DH] planar head layout.
// mode 1: write fp32 interleaved complex64: out[2*(m*Nd+n)+comp].
// negmask (0x80008000) negates the A2 pass (bf16 sign flip while staging).
// ---------------------------------------------------------------------------
__global__ __launch_bounds__(256) void k_gemm_bf16(
    const bf16* __restrict__ A1, const bf16* __restrict__ W1,
    const bf16* __restrict__ A2, const bf16* __restrict__ W2,
    unsigned int negmask, const float* __restrict__ bias,
    float* __restrict__ out, int mode, int comp) {
  const int K = kD, Nd = kD;
  __shared__ bf16 ldsA[2][128 * 32];
  __shared__ bf16 ldsB[2][128 * 32];

  int bid = blockIdx.x;
  const int nTilesN = Nd / 128;        // 8
  int bm = bid / nTilesN;
  int bn = bid % nTilesN;
  int t = threadIdx.x;
  int lane = t & 31, wid = t >> 5;
  int wm = wid & 3;                    // 4 row-groups of 32
  int wn = wid >> 2;                   // 2 col-groups of 64
  int lr = lane & 15;
  int hi = lane >> 4;

  v8f acc[2][4];
#pragma unroll
  for (int mi = 0; mi < 2; ++mi)
#pragma unroll
    for (int ni = 0; ni < 4; ++ni) acc[mi][ni] = v8f_zero();

  int srow = t >> 1;                   // staging row 0..127
  int shalf = t & 1;                   // 16-element half

  const int nSteps = (A2 != nullptr) ? 64 : 32;   // flattened dual-pass K loop
  uint4 ra0, ra1, rb0, rb1;                        // register-staged tile

  auto ld_tile = [&](int step) {
    const bf16* A = (step < 32) ? A1 : A2;
    const bf16* W = (step < 32) ? W1 : W2;
    unsigned int nm = (step < 32) ? 0u : negmask;
    int k0 = (step & 31) * 32;
    const uint4* pa = (const uint4*)(A + (size_t)(bm * 128 + srow) * K + k0 + shalf * 16);
    ra0 = pa[0]; ra1 = pa[1];
    if (nm) {
      ra0.x ^= nm; ra0.y ^= nm; ra0.z ^= nm; ra0.w ^= nm;
      ra1.x ^= nm; ra1.y ^= nm; ra1.z ^= nm; ra1.w ^= nm;
    }
    const uint4* pb = (const uint4*)(W + (size_t)(bn * 128 + srow) * K + k0 + shalf * 16);
    rb0 = pb[0]; rb1 = pb[1];
  };
  auto st_tile = [&](int buf) {
    uint4* da = (uint4*)&ldsA[buf][srow * 32 + shalf * 16];
    da[0] = ra0; da[1] = ra1;
    uint4* db = (uint4*)&ldsB[buf][srow * 32 + shalf * 16];
    db[0] = rb0; db[1] = rb1;
  };

  // prologue: tile 0 into buffer 0
  ld_tile(0);
  st_tile(0);
  __syncthreads();

  for (int s = 0; s < nSteps; ++s) {
    int buf = s & 1;
    if (s + 1 < nSteps) ld_tile(s + 1);       // global loads overlap compute

    // ---- load ALL fragments first (single dscnt wait), then 8 WMMAs ----
    FragAB a[2], b[4];
#pragma unroll
    for (int mi = 0; mi < 2; ++mi) {
      int Mrow = wm * 32 + mi * 16 + lr;
#pragma unroll
      for (int j = 0; j < 8; ++j) {
        int kk = (j < 4) ? (2 * j + hi * 8) : (16 + 2 * (j - 4) + hi * 8);
        a[mi].u[j] = *(const unsigned int*)&ldsA[buf][Mrow * 32 + kk];
      }
    }
#pragma unroll
    for (int ni = 0; ni < 4; ++ni) {
      int Ncol = wn * 64 + ni * 16 + lr;
#pragma unroll
      for (int j = 0; j < 8; ++j) {
        int kk = (j < 4) ? (2 * j + hi * 8) : (16 + 2 * (j - 4) + hi * 8);
        b[ni].u[j] = *(const unsigned int*)&ldsB[buf][Ncol * 32 + kk];
      }
    }
#pragma unroll
    for (int ni = 0; ni < 4; ++ni) {
      acc[0][ni] = WMMA_BF16(a[0].v, b[ni].v, acc[0][ni]);
      acc[1][ni] = WMMA_BF16(a[1].v, b[ni].v, acc[1][ni]);
    }

    if (s + 1 < nSteps) {
      st_tile(buf ^ 1);                       // fill the other buffer
      __syncthreads();                        // publish before next compute
    }
  }

  // ---- epilogue ----
#pragma unroll
  for (int mi = 0; mi < 2; ++mi)
#pragma unroll
    for (int ni = 0; ni < 4; ++ni)
#pragma unroll
      for (int r = 0; r < 8; ++r) {
        int Mg = bm * 128 + wm * 32 + mi * 16 + r + hi * 8;
        int Ng = bn * 128 + wn * 64 + ni * 16 + lr;
        float val = acc[mi][ni][r] + bias[Ng];
        if (mode == 0) {
          int bb = Mg >> 10, tok = Mg & 1023;
          int h = Ng >> 6, dh = Ng & 63;
          out[(size_t)((((bb << 4) + h) << 10) + tok) * 64 + dh] = val;
        } else {
          out[2 * ((size_t)Mg * Nd + Ng) + comp] = val;
        }
      }
}

// ---------------------------------------------------------------------------
// Fused flash attention over FFT'd q/k/v.  One wave = 16 query rows of one
// (b,h).  Scores: S = qf . conj(kf)^H decomposed into real bf16 WMMAs; softmax
// over |S| with running max/sum; PV via LDS-reshaped probability tile.
// V fragments are loaded up-front each iteration so their global latency hides
// under the score WMMAs + softmax VALU chain.
// ---------------------------------------------------------------------------
__global__ __launch_bounds__(256) void k_attn(
    const bf16* __restrict__ qr, const bf16* __restrict__ qi,
    const bf16* __restrict__ kr, const bf16* __restrict__ ki,
    const bf16* __restrict__ vrT, const bf16* __restrict__ viT,
    float* __restrict__ outr, float* __restrict__ outi) {
  __shared__ bf16 pbuf[8][16 * 32];     // per-wave probability tile

  int lane = threadIdx.x & 31;
  int wid = threadIdx.x >> 5;
  int gw = blockIdx.x * 8 + wid;        // 0 .. 2047
  int bh = gw >> 6;                     // 0 .. 31
  int m0 = (gw & 63) * 16;              // query row base
  const size_t offQK = (size_t)bh * (kN * kDH);
  const size_t offVT = (size_t)bh * (kDH * kN);
  int lr = lane & 15;
  int hi = lane >> 4;

  // q A-fragments, kept live for the whole row block
  FragAB aqr[2], aqi[2];
#pragma unroll
  for (int ks = 0; ks < 2; ++ks) {
    int Mrow = m0 + lr;
#pragma unroll
    for (int j = 0; j < 8; ++j) {
      int kk = ks * 32 + ((j < 4) ? (2 * j + hi * 8) : (16 + 2 * (j - 4) + hi * 8));
      aqr[ks].u[j] = *(const unsigned int*)(qr + offQK + (size_t)Mrow * 64 + kk);
      aqi[ks].u[j] = *(const unsigned int*)(qi + offQK + (size_t)Mrow * 64 + kk);
    }
  }

  v8f accR[4], accI[4];
#pragma unroll
  for (int dt = 0; dt < 4; ++dt) { accR[dt] = v8f_zero(); accI[dt] = v8f_zero(); }
  float mrun[8], lrun[8];
#pragma unroll
  for (int r = 0; r < 8; ++r) { mrun[r] = -3.0e38f; lrun[r] = 0.0f; }

  for (int j0 = 0; j0 < kN; j0 += 32) {
    // ---- V fragments for this key block: independent, issue first ----
    FragAB bvr[4], bvi[4];
#pragma unroll
    for (int dt = 0; dt < 4; ++dt) {
      int d = dt * 16 + lr;
#pragma unroll
      for (int j = 0; j < 8; ++j) {
        int kj = j0 + ((j < 4) ? (2 * j + hi * 8) : (16 + 2 * (j - 4) + hi * 8));
        bvr[dt].u[j] = *(const unsigned int*)(vrT + offVT + (size_t)d * kN + kj);
        bvi[dt].u[j] = *(const unsigned int*)(viT + offVT + (size_t)d * kN + kj);
      }
    }

    // ---- complex scores for two 16-key subtiles ----
    float mag[2][8];
#pragma unroll
    for (int c = 0; c < 2; ++c) {
      int cb = j0 + c * 16;
      FragAB bkr[2], bki[2], bkiN[2];
#pragma unroll
      for (int ks = 0; ks < 2; ++ks) {
        int key = cb + lr;
#pragma unroll
        for (int j = 0; j < 8; ++j) {
          int kk = ks * 32 + ((j < 4) ? (2 * j + hi * 8) : (16 + 2 * (j - 4) + hi * 8));
          bkr[ks].u[j] = *(const unsigned int*)(kr + offQK + (size_t)key * 64 + kk);
          unsigned int u = *(const unsigned int*)(ki + offQK + (size_t)key * 64 + kk);
          bki[ks].u[j] = u;
          bkiN[ks].u[j] = u ^ 0x80008000u;   // -ki (bf16 WMMA has no A/B negate)
        }
      }
      v8f sre = v8f_zero(), sim = v8f_zero();
      // Re(S) = qr.kr + qi.ki
      sre = WMMA_BF16(aqr[0].v, bkr[0].v, sre);
      sre = WMMA_BF16(aqr[1].v, bkr[1].v, sre);
      sre = WMMA_BF16(aqi[0].v, bki[0].v, sre);
      sre = WMMA_BF16(aqi[1].v, bki[1].v, sre);
      // Im(S) = qi.kr - qr.ki
      sim = WMMA_BF16(aqi[0].v, bkr[0].v, sim);
      sim = WMMA_BF16(aqi[1].v, bkr[1].v, sim);
      sim = WMMA_BF16(aqr[0].v, bkiN[0].v, sim);
      sim = WMMA_BF16(aqr[1].v, bkiN[1].v, sim);
#pragma unroll
      for (int r = 0; r < 8; ++r)
        mag[c][r] = __builtin_sqrtf(sre[r] * sre[r] + sim[r] * sim[r]);
    }

    // online softmax statistics (rows live in 16-lane halves, reg index = row)
    float tm[8];
#pragma unroll
    for (int r = 0; r < 8; ++r) tm[r] = fmaxf(mag[0][r], mag[1][r]);
#pragma unroll
    for (int d = 1; d < 16; d <<= 1)
#pragma unroll
      for (int r = 0; r < 8; ++r) tm[r] = fmaxf(tm[r], __shfl_xor(tm[r], d, 16));

    float alpha[8];
#pragma unroll
    for (int r = 0; r < 8; ++r) {
      float nm = fmaxf(mrun[r], tm[r]);
      alpha[r] = __expf(mrun[r] - nm);
      mrun[r] = nm;
    }
    float p0[8], p1[8], ps[8];
#pragma unroll
    for (int r = 0; r < 8; ++r) {
      p0[r] = __expf(mag[0][r] - mrun[r]);
      p1[r] = __expf(mag[1][r] - mrun[r]);
      ps[r] = p0[r] + p1[r];
    }
#pragma unroll
    for (int d = 1; d < 16; d <<= 1)
#pragma unroll
      for (int r = 0; r < 8; ++r) ps[r] += __shfl_xor(ps[r], d, 16);
#pragma unroll
    for (int r = 0; r < 8; ++r) lrun[r] = lrun[r] * alpha[r] + ps[r];
#pragma unroll
    for (int dt = 0; dt < 4; ++dt)
#pragma unroll
      for (int r = 0; r < 8; ++r) { accR[dt][r] *= alpha[r]; accI[dt][r] *= alpha[r]; }

    // C-layout -> LDS [16 rows][32 cols] bf16 -> reload as A fragment
    bf16* pl = pbuf[wid];
#pragma unroll
    for (int r = 0; r < 8; ++r) {
      int row = r + hi * 8;
      pl[row * 32 + lr] = (bf16)p0[r];
      pl[row * 32 + 16 + lr] = (bf16)p1[r];
    }
    FragAB pA;
    {
      int Mrow = lr;
#pragma unroll
      for (int j = 0; j < 8; ++j) {
        int kk = (j < 4) ? (2 * j + hi * 8) : (16 + 2 * (j - 4) + hi * 8);
        pA.u[j] = *(const unsigned int*)&pl[Mrow * 32 + kk];
      }
    }
    // PV: 32 keys x 64 dh (V fragments already resident)
#pragma unroll
    for (int dt = 0; dt < 4; ++dt) {
      accR[dt] = WMMA_BF16(pA.v, bvr[dt].v, accR[dt]);
      accI[dt] = WMMA_BF16(pA.v, bvi[dt].v, accI[dt]);
    }
  }

  // normalize + store (fp32 planar [B,H,N,DH])
  float inv[8];
#pragma unroll
  for (int r = 0; r < 8; ++r) inv[r] = 1.0f / lrun[r];
#pragma unroll
  for (int dt = 0; dt < 4; ++dt)
#pragma unroll
    for (int r = 0; r < 8; ++r) {
      int Mrow = m0 + r + hi * 8;
      int d = dt * 16 + lr;
      outr[offQK + (size_t)Mrow * 64 + d] = accR[dt][r] * inv[r];
      outi[offQK + (size_t)Mrow * 64 + d] = accI[dt][r] * inv[r];
    }
}

// ---------------------------------------------------------------------------
// Host-side orchestration
// ---------------------------------------------------------------------------
extern "C" void kernel_launch(void* const* d_in, const int* in_sizes, int n_in,
                              void* d_out, int out_size, void* d_ws, size_t ws_size,
                              hipStream_t stream) {
  const float* x   = (const float*)d_in[0];
  const float* Wsrc[8] = {
      (const float*)d_in[1],  (const float*)d_in[2],   // Wqr, Wqi
      (const float*)d_in[5],  (const float*)d_in[6],   // Wkr, Wki
      (const float*)d_in[9],  (const float*)d_in[10],  // Wvr, Wvi
      (const float*)d_in[13], (const float*)d_in[14]}; // Wor, Woi
  const float* bqr = (const float*)d_in[3];
  const float* bqi = (const float*)d_in[4];
  const float* bkr = (const float*)d_in[7];
  const float* bki = (const float*)d_in[8];
  const float* bvr = (const float*)d_in[11];
  const float* bvi = (const float*)d_in[12];
  const float* bor = (const float*)d_in[15];
  const float* boi = (const float*)d_in[16];

  char* ws = (char*)d_ws;
  size_t off = 0;
  auto carve = [&](size_t bytes) -> void* {
    void* p = ws + off;
    off = (off + bytes + 255) & ~(size_t)255;
    return p;
  };

  bf16* x_bf = (bf16*)carve((size_t)kM * kD * 2);
  bf16* Wbf[8];
  for (int i = 0; i < 8; ++i) Wbf[i] = (bf16*)carve((size_t)kD * kD * 2);
  float* q_r = (float*)carve((size_t)kBHND * 4);
  float* q_i = (float*)carve((size_t)kBHND * 4);
  float* k_r = (float*)carve((size_t)kBHND * 4);
  float* k_i = (float*)carve((size_t)kBHND * 4);
  float* v_r = (float*)carve((size_t)kBHND * 4);
  float* v_i = (float*)carve((size_t)kBHND * 4);
  bf16* qr_bf = (bf16*)carve((size_t)kBHND * 2);
  bf16* qi_bf = (bf16*)carve((size_t)kBHND * 2);
  bf16* kr_bf = (bf16*)carve((size_t)kBHND * 2);
  bf16* ki_bf = (bf16*)carve((size_t)kBHND * 2);
  bf16* vrT_bf = (bf16*)carve((size_t)kBHND * 2);
  bf16* viT_bf = (bf16*)carve((size_t)kBHND * 2);
  float* outf_r = (float*)carve((size_t)kBHND * 4);
  float* outf_i = (float*)carve((size_t)kBHND * 4);
  bf16* o_r = (bf16*)carve((size_t)kBHND * 2);
  bf16* o_i = (bf16*)carve((size_t)kBHND * 2);
  (void)ws_size; (void)in_sizes; (void)n_in; (void)out_size;

  // 1) bf16 conversions of x and weights
  k_cvt_bf16<<<(kM * kD) / 256, 256, 0, stream>>>(x, x_bf, kM * kD);
  for (int i = 0; i < 8; ++i)
    k_cvt_bf16<<<(kD * kD) / 256, 256, 0, stream>>>(Wsrc[i], Wbf[i], kD * kD);

  // 2) Q/K/V projections (real input -> re/im are independent real GEMMs)
  k_gemm_bf16<<<128, 256, 0, stream>>>(x_bf, Wbf[0], nullptr, nullptr, 0u, bqr, q_r, 0, 0);
  k_gemm_bf16<<<128, 256, 0, stream>>>(x_bf, Wbf[1], nullptr, nullptr, 0u, bqi, q_i, 0, 0);
  k_gemm_bf16<<<128, 256, 0, stream>>>(x_bf, Wbf[2], nullptr, nullptr, 0u, bkr, k_r, 0, 0);
  k_gemm_bf16<<<128, 256, 0, stream>>>(x_bf, Wbf[3], nullptr, nullptr, 0u, bki, k_i, 0, 0);
  k_gemm_bf16<<<128, 256, 0, stream>>>(x_bf, Wbf[4], nullptr, nullptr, 0u, bvr, v_r, 0, 0);
  k_gemm_bf16<<<128, 256, 0, stream>>>(x_bf, Wbf[5], nullptr, nullptr, 0u, bvi, v_i, 0, 0);

  // 3) forward FFTs along the sequence axis (in place, ortho)
  k_fft1024<<<kB * kH * kDH, 256, 0, stream>>>(q_r, q_i, -1.0f);
  k_fft1024<<<kB * kH * kDH, 256, 0, stream>>>(k_r, k_i, -1.0f);
  k_fft1024<<<kB * kH * kDH, 256, 0, stream>>>(v_r, v_i, -1.0f);

  // 4) bf16 copies for the attention WMMAs (vf transposed to [BH][DH][N])
  k_cvt_bf16<<<kBHND / 256, 256, 0, stream>>>(q_r, qr_bf, kBHND);
  k_cvt_bf16<<<kBHND / 256, 256, 0, stream>>>(q_i, qi_bf, kBHND);
  k_cvt_bf16<<<kBHND / 256, 256, 0, stream>>>(k_r, kr_bf, kBHND);
  k_cvt_bf16<<<kBHND / 256, 256, 0, stream>>>(k_i, ki_bf, kBHND);
  k_cvt_vfT<<<kBHND / 256, 256, 0, stream>>>(v_r, vrT_bf);
  k_cvt_vfT<<<kBHND / 256, 256, 0, stream>>>(v_i, viT_bf);

  // 5) fused flash attention on complex spectra
  k_attn<<<256, 256, 0, stream>>>(qr_bf, qi_bf, kr_bf, ki_bf, vrT_bf, viT_bf,
                                  outf_r, outf_i);

  // 6) inverse FFT (in place, ortho)
  k_fft1024<<<kB * kH * kDH, 256, 0, stream>>>(outf_r, outf_i, 1.0f);

  // 7) back to [B,N,D] bf16 A-matrices
  k_cvt_outT<<<kBHND / 256, 256, 0, stream>>>(outf_r, o_r);
  k_cvt_outT<<<kBHND / 256, 256, 0, stream>>>(outf_i, o_i);

  // 8) output complex linear -> interleaved complex64 in d_out
  float* out = (float*)d_out;
  // yr = o_r @ Wor^T - o_i @ Woi^T + bor
  k_gemm_bf16<<<128, 256, 0, stream>>>(o_r, Wbf[6], o_i, Wbf[7], 0x80008000u, bor, out, 1, 0);
  // yi = o_r @ Woi^T + o_i @ Wor^T + boi
  k_gemm_bf16<<<128, 256, 0, stream>>>(o_r, Wbf[7], o_i, Wbf[6], 0u, boi, out, 1, 1);
}
